// GIN0_52037823758418
// MI455X (gfx1250) — matrix operand order repaired and checked by reading, not compiled, not run on hardware
//
#include <hip/hip_runtime.h>
#include <hip/hip_bf16.h>

typedef _Float16 half_t;
typedef __attribute__((ext_vector_type(16))) _Float16 v16h;
typedef __attribute__((ext_vector_type(8)))  float    v8f;

#define LSTRIDE 68   // LDS row stride in floats (avoids 2-way conflicts between lane halves)
#define WAVES   8    // waves per block (256 threads)

// ---------------------------------------------------------------------------
// WMMA helpers
// ---------------------------------------------------------------------------
__device__ __forceinline__ v8f wmma_f16(v16h a, v16h b, v8f c) {
    // D = A(16x32 f16) * B(32x16 f16) + C(16x16 f32)
    return __builtin_amdgcn_wmma_f32_16x16x32_f16(
        /*neg_a=*/false, a, /*neg_b=*/false, b,
        /*c_mod=*/(short)0, c, /*reuse_a=*/false, /*reuse_b=*/false);
}

// Load a 16x32 f32 tile (row-major, given row stride) into the CDNA5 16-bit
// A-matrix register layout:
//   lanes 0-15 : row M = lane,   a[0..7] = K k0+0..7,  a[8..15] = K k0+16..23
//   lanes 16-31: row M = lane-16,a[0..7] = K k0+8..15, a[8..15] = K k0+24..31
__device__ __forceinline__ v16h load_a_tile(const float* __restrict__ base,
                                            int stride, int lane, int k0) {
    const float* row = base + (lane & 15) * stride;
    int kb = k0 + ((lane >> 4) << 3);
    float4 x0 = *(const float4*)(row + kb);
    float4 x1 = *(const float4*)(row + kb + 4);
    float4 y0 = *(const float4*)(row + kb + 16);
    float4 y1 = *(const float4*)(row + kb + 20);
    v16h a;
    a[0]  = (half_t)x0.x; a[1]  = (half_t)x0.y; a[2]  = (half_t)x0.z; a[3]  = (half_t)x0.w;
    a[4]  = (half_t)x1.x; a[5]  = (half_t)x1.y; a[6]  = (half_t)x1.z; a[7]  = (half_t)x1.w;
    a[8]  = (half_t)y0.x; a[9]  = (half_t)y0.y; a[10] = (half_t)y0.z; a[11] = (half_t)y0.w;
    a[12] = (half_t)y1.x; a[13] = (half_t)y1.y; a[14] = (half_t)y1.z; a[15] = (half_t)y1.w;
    return a;
}

// ---------------------------------------------------------------------------
// Weight pre-swizzle: 11 matrices (64x64, W[k*64+n]) -> B-matrix layout tiles.
// One wave per (mat, kstep, ntile). Tile = 512 f16: lane*16 contiguous halves:
//   lanes 0-15 : col n = nt*16+lane,     K = ks*32 + 0..7, 16..23
//   lanes 16-31: col n = nt*16+lane-16,  K = ks*32 + 8..15, 24..31
// ---------------------------------------------------------------------------
__global__ void gin_prep_weights(const float* __restrict__ c1W1, const float* __restrict__ c1W2,
                                 const float* __restrict__ csW1, const float* __restrict__ csW2,
                                 const float* __restrict__ linW, half_t* __restrict__ wbuf) {
    int blk  = blockIdx.x;        // 0..87
    int lane = threadIdx.x;       // 0..31
    int mat  = blk >> 3;          // 0..10
    int ks   = (blk >> 2) & 1;
    int nt   = blk & 3;
    const float* W;
    if      (mat == 0)  W = c1W1;
    else if (mat == 1)  W = c1W2;
    else if (mat < 6)   W = csW1 + (mat - 2) * 4096;
    else if (mat < 10)  W = csW2 + (mat - 6) * 4096;
    else                W = linW;
    int n  = nt * 16 + (lane & 15);
    int kb = ks * 32 + ((lane >> 4) << 3);
    half_t* dst = wbuf + (size_t)blk * 512 + lane * 16;
#pragma unroll
    for (int i = 0; i < 16; ++i) dst[i] = (half_t)W[(kb + i + (i >= 8 ? 8 : 0)) * 64 + n];
}

// Fold BatchNorm (eval) into scale/shift per channel, 5 layers x 64 channels.
__global__ void gin_prep_bn(const float* __restrict__ c1_g, const float* __restrict__ c1_be,
                            const float* __restrict__ c1_m, const float* __restrict__ c1_v,
                            const float* __restrict__ cs_g, const float* __restrict__ cs_be,
                            const float* __restrict__ cs_m, const float* __restrict__ cs_v,
                            float* __restrict__ scale, float* __restrict__ shift) {
    int i = blockIdx.x * 64 + threadIdx.x;  // 0..319
    if (i >= 320) return;
    int layer = i >> 6, c = i & 63;
    float g, be, m, v;
    if (layer == 0) { g = c1_g[c]; be = c1_be[c]; m = c1_m[c]; v = c1_v[c]; }
    else {
        int l = layer - 1;
        g = cs_g[l * 64 + c]; be = cs_be[l * 64 + c];
        m = cs_m[l * 64 + c]; v = cs_v[l * 64 + c];
    }
    float s = g * rsqrtf(v + 1e-5f);
    scale[i] = s;
    shift[i] = be - m * s;
}

// z = src (node features), zero-padded to the 16-row tile boundary.
__global__ void gin_copy_pad(const float* __restrict__ src, float* __restrict__ dst,
                             int valid, int total4) {
    int i = blockIdx.x * 256 + threadIdx.x;
    if (i >= total4) return;
    int base = i * 4;
    float4 v;
    if (base + 3 < valid) v = ((const float4*)src)[i];
    else {
        float t[4];
#pragma unroll
        for (int j = 0; j < 4; ++j) t[j] = (base + j < valid) ? src[base + j] : 0.f;
        v = make_float4(t[0], t[1], t[2], t[3]);
    }
    ((float4*)dst)[i] = v;
}

// z[dst] += h[src] for every edge; one thread = (edge, 4 features).
__global__ void gin_scatter_add(const int* __restrict__ src, const int* __restrict__ dst,
                                const float* __restrict__ h, float* __restrict__ z, int total) {
    int i = blockIdx.x * 256 + threadIdx.x;
    if (i >= total) return;
    int e = i >> 4;
    int f = (i & 15) * 4;
    int s = src[e], d = dst[e];
    float4 v = *(const float4*)(h + (size_t)s * 64 + f);
    float* zp = z + (size_t)d * 64 + f;
    atomicAdd(zp + 0, v.x);
    atomicAdd(zp + 1, v.y);
    atomicAdd(zp + 2, v.z);
    atomicAdd(zp + 3, v.w);
}

// ---------------------------------------------------------------------------
// GIN MLP: h_out = [h_in +] BN(relu(relu(z@W1+b1)@W2+b2))
// One wave per 16-node tile; 4 N-tiles x 2 K-steps per GEMM; t1 staged in LDS.
// ---------------------------------------------------------------------------
__global__ __launch_bounds__(256) void gin_mlp(
    const float* __restrict__ z, const float* __restrict__ h_in, float* __restrict__ h_out,
    const half_t* __restrict__ w1, const half_t* __restrict__ w2,
    const float* __restrict__ b1, const float* __restrict__ b2,
    const float* __restrict__ scale, const float* __restrict__ shift,
    int residual, int ntiles) {
    __shared__ __align__(16) float t1[WAVES][16][LSTRIDE];
    int wave = threadIdx.x >> 5;
    int lane = threadIdx.x & 31;
    int tile = blockIdx.x * WAVES + wave;
    if (tile >= ntiles) return;  // wave-uniform: EXEC stays all-1s for WMMA

    const float* ztile = z + (size_t)tile * 1024;
    v16h a0 = load_a_tile(ztile, 64, lane, 0);
    v16h a1 = load_a_tile(ztile, 64, lane, 32);

    int col_lo = lane & 15;
    int rowsel = (lane >> 4) << 3;  // 0 or 8: D-layout row offset

    // GEMM1 + bias + ReLU -> LDS (row-major 16x64, stride LSTRIDE)
#pragma unroll
    for (int nt = 0; nt < 4; ++nt) {
        v16h bb0 = *(const v16h*)(w1 + nt * 512 + lane * 16);
        v16h bb1 = *(const v16h*)(w1 + (4 + nt) * 512 + lane * 16);
        v8f c = {};
        c = wmma_f16(a0, bb0, c);
        c = wmma_f16(a1, bb1, c);
        int col = nt * 16 + col_lo;
        float bias = b1[col];
#pragma unroll
        for (int v = 0; v < 8; ++v) {
            float val = c[v] + bias;
            t1[wave][v + rowsel][col] = val > 0.f ? val : 0.f;
        }
    }
    // same-wave LDS RAW: drain DS ops + stop compiler reordering
    asm volatile("s_wait_dscnt 0" ::: "memory");

    v16h d0 = load_a_tile(&t1[wave][0][0], LSTRIDE, lane, 0);
    v16h d1 = load_a_tile(&t1[wave][0][0], LSTRIDE, lane, 32);

    // GEMM2 + bias + ReLU + BN (+ residual) -> global
#pragma unroll
    for (int nt = 0; nt < 4; ++nt) {
        v16h bb0 = *(const v16h*)(w2 + nt * 512 + lane * 16);
        v16h bb1 = *(const v16h*)(w2 + (4 + nt) * 512 + lane * 16);
        v8f c = {};
        c = wmma_f16(d0, bb0, c);
        c = wmma_f16(d1, bb1, c);
        int col = nt * 16 + col_lo;
        float bias = b2[col], sc = scale[col], sh = shift[col];
#pragma unroll
        for (int v = 0; v < 8; ++v) {
            float val = c[v] + bias;
            val = val > 0.f ? val : 0.f;
            val = val * sc + sh;
            size_t idx = ((size_t)tile * 16 + v + rowsel) * 64 + col;
            h_out[idx] = (residual ? h_in[idx] : 0.f) + val;  // in-place safe (own tile only)
        }
    }
}

// Head: out = tanh(h @ lin_W + lin_b)
__global__ __launch_bounds__(256) void gin_head(
    const float* __restrict__ h, float* __restrict__ out,
    const half_t* __restrict__ wl, const float* __restrict__ lb, int ntiles) {
    int wave = threadIdx.x >> 5;
    int lane = threadIdx.x & 31;
    int tile = blockIdx.x * WAVES + wave;
    if (tile >= ntiles) return;
    const float* ht = h + (size_t)tile * 1024;
    v16h a0 = load_a_tile(ht, 64, lane, 0);
    v16h a1 = load_a_tile(ht, 64, lane, 32);
    int col_lo = lane & 15;
    int rowsel = (lane >> 4) << 3;
#pragma unroll
    for (int nt = 0; nt < 4; ++nt) {
        v16h bb0 = *(const v16h*)(wl + nt * 512 + lane * 16);
        v16h bb1 = *(const v16h*)(wl + (4 + nt) * 512 + lane * 16);
        v8f c = {};
        c = wmma_f16(a0, bb0, c);
        c = wmma_f16(a1, bb1, c);
        int col = nt * 16 + col_lo;
        float bias = lb[col];
#pragma unroll
        for (int v = 0; v < 8; ++v)
            out[((size_t)tile * 16 + v + rowsel) * 64 + col] = tanhf(c[v] + bias);
    }
}

// Per-graph node counts via binary search on the sorted batch array.
__global__ void gin_count(const int* __restrict__ batch, float* __restrict__ cnt, int N, int G) {
    int g = blockIdx.x * blockDim.x + threadIdx.x;
    if (g >= G) return;
    int lo = 0, hi = N;
    while (lo < hi) { int mid = (lo + hi) >> 1; if (batch[mid] < g) lo = mid + 1; else hi = mid; }
    int a = lo;
    lo = 0; hi = N;
    while (lo < hi) { int mid = (lo + hi) >> 1; if (batch[mid] < g + 1) lo = mid + 1; else hi = mid; }
    cnt[g] = (float)(lo - a);
}

// Segment sums: thread = (32-node chunk, feature); run-length fold before atomics.
__global__ void gin_pool(const float* __restrict__ out, const int* __restrict__ batch,
                         float* __restrict__ pool, int N, int nchunks) {
    int t = blockIdx.x * 256 + threadIdx.x;
    int f = t & 63;
    int chunk = t >> 6;
    if (chunk >= nchunks) return;
    int i = chunk * 32;
    int end = i + 32; if (end > N) end = N;
    int curb = batch[i];
    float acc = 0.f;
    for (; i < end; ++i) {
        int b = batch[i];
        if (b != curb) { atomicAdd(&pool[curb * 64 + f], acc); acc = 0.f; curb = b; }
        acc += out[(size_t)i * 64 + f];
    }
    atomicAdd(&pool[curb * 64 + f], acc);
}

__global__ void gin_div(const float* __restrict__ pool, const float* __restrict__ cnt,
                        float* __restrict__ out, int total) {
    int i = blockIdx.x * 256 + threadIdx.x;
    if (i >= total) return;
    out[i] = pool[i] / fmaxf(cnt[i >> 6], 1.f);
}

// ---------------------------------------------------------------------------
extern "C" void kernel_launch(void* const* d_in, const int* in_sizes, int n_in,
                              void* d_out, int out_size, void* d_ws, size_t ws_size,
                              hipStream_t stream) {
    const float* x     = (const float*)d_in[0];
    const int*   ei    = (const int*)  d_in[1];
    const int*   batch = (const int*)  d_in[2];
    const float* c1_W1 = (const float*)d_in[3];
    const float* c1_b1 = (const float*)d_in[4];
    const float* c1_W2 = (const float*)d_in[5];
    const float* c1_b2 = (const float*)d_in[6];
    const float* c1_g  = (const float*)d_in[7];
    const float* c1_be = (const float*)d_in[8];
    const float* c1_m  = (const float*)d_in[9];
    const float* c1_v  = (const float*)d_in[10];
    const float* cs_W1 = (const float*)d_in[11];
    const float* cs_b1 = (const float*)d_in[12];
    const float* cs_W2 = (const float*)d_in[13];
    const float* cs_b2 = (const float*)d_in[14];
    const float* cs_g  = (const float*)d_in[15];
    const float* cs_be = (const float*)d_in[16];
    const float* cs_m  = (const float*)d_in[17];
    const float* cs_v  = (const float*)d_in[18];
    const float* lin_W = (const float*)d_in[19];
    const float* lin_b = (const float*)d_in[20];

    int N = in_sizes[0] / 64;
    int E = in_sizes[1] / 2;
    int G = out_size / 64;
    int ntiles = (N + 15) / 16;
    int Npad   = ntiles * 16;

    // workspace carve-up
    char*   ws      = (char*)d_ws;
    float*  h       = (float*)ws;                                   // Npad*64 f32
    float*  z       = (float*)(ws + (size_t)Npad * 256);            // Npad*64 f32 (also head output)
    char*   tail    = ws + 2 * (size_t)Npad * 256;
    half_t* wbuf    = (half_t*)tail;                                // 88*512 f16
    float*  bnscale = (float*)(tail + 88 * 512 * 2);                // 320 f32
    float*  bnshift = bnscale + 320;                                // 320 f32
    float*  pool    = bnshift + 320;                                // G*64 f32
    float*  cnt     = pool + (size_t)G * 64;                        // G f32
    (void)n_in; (void)ws_size;

    const int* esrc = ei;
    const int* edst = ei + E;

    gin_prep_weights<<<88, 32, 0, stream>>>(c1_W1, c1_W2, cs_W1, cs_W2, lin_W, wbuf);
    gin_prep_bn<<<5, 64, 0, stream>>>(c1_g, c1_be, c1_m, c1_v, cs_g, cs_be, cs_m, cs_v,
                                      bnscale, bnshift);

    int total4     = Npad * 16;                       // float4 copies
    int copyBlocks = (total4 + 255) / 256;
    int scatTotal  = E * 16;                          // (edge, 4-feature) threads
    int scatBlocks = (scatTotal + 255) / 256;
    int mlpBlocks  = (ntiles + WAVES - 1) / WAVES;

    // conv1: z = x + agg(x); h = MLP_BN(z)
    gin_copy_pad<<<copyBlocks, 256, 0, stream>>>(x, z, N * 64, total4);
    gin_scatter_add<<<scatBlocks, 256, 0, stream>>>(esrc, edst, x, z, scatTotal);
    gin_mlp<<<mlpBlocks, 256, 0, stream>>>(z, h, h, wbuf, wbuf + 4096,
                                           c1_b1, c1_b2, bnscale, bnshift, 0, ntiles);

    // residual layers: h = h + MLP_BN(h + agg(h))
    for (int l = 0; l < 4; ++l) {
        gin_copy_pad<<<copyBlocks, 256, 0, stream>>>(h, z, N * 64, total4);
        gin_scatter_add<<<scatBlocks, 256, 0, stream>>>(esrc, edst, h, z, scatTotal);
        gin_mlp<<<mlpBlocks, 256, 0, stream>>>(z, h, h,
                                               wbuf + (size_t)(2 + l) * 4096,
                                               wbuf + (size_t)(6 + l) * 4096,
                                               cs_b1 + l * 64, cs_b2 + l * 64,
                                               bnscale + (l + 1) * 64, bnshift + (l + 1) * 64,
                                               1, ntiles);
    }

    // head: out(z) = tanh(h @ lin_W + lin_b)
    gin_head<<<mlpBlocks, 256, 0, stream>>>(h, z, wbuf + 10 * 4096, lin_b, ntiles);

    // global mean pool
    hipMemsetAsync(pool, 0, (size_t)G * 64 * sizeof(float), stream);
    gin_count<<<(G + 127) / 128, 128, 0, stream>>>(batch, cnt, N, G);
    int nchunks = (N + 31) / 32;
    int poolBlocks = (nchunks * 64 + 255) / 256;
    gin_pool<<<poolBlocks, 256, 0, stream>>>(z, batch, pool, N, nchunks);
    gin_div<<<(G * 64 + 255) / 256, 256, 0, stream>>>(pool, cnt, (float*)d_out, G * 64);
}
